// Attention_nl_25812753449030
// MI455X (gfx1250) — matrix operand order repaired and checked by reading, not compiled, not run on hardware
//
#include <hip/hip_runtime.h>

// ---------------------------------------------------------------------------
// Problem constants (B,C,H,W) = (8,256,48,48), N = H*W = 2304
// ---------------------------------------------------------------------------
#define BB 8
#define CC 256
#define NN 2304
#define BN_EPS 1e-5f

typedef __bf16 bf16;
typedef __attribute__((ext_vector_type(16))) __bf16 v16bf;
typedef __attribute__((ext_vector_type(8)))  __bf16 v8bf;
typedef __attribute__((ext_vector_type(8)))  float  v8f;

union V16 { v16bf v; v8bf h[2]; };

// Output modes for the generic GEMM
#define OUT_F32_N  0   // f32, row-major M x N (+ optional bias[m])
#define OUT_BF16_T 1   // bf16, transposed (N x M) -- single b128 store/lane
#define OUT_BF16_N 2   // bf16, row-major M x N

// ---------------------------------------------------------------------------
// Generic bf16 WMMA GEMM:  D(MxN) = A(MxK) * B(KxN),  B given transposed (NxK)
// Block = 128 threads = 4 waves arranged 2(M) x 2(N).
// Each wave owns a 32x64 tile (2x4 of 16x16 WMMA tiles) -> block tile 64x128.
// Per K=32 step: 8 WMMAs vs 6x32B lane loads (1.33x the FLOP/byte of 32x32).
// grid = (N/128, M/64, batch)
// ---------------------------------------------------------------------------
template<int MODE>
__global__ __launch_bounds__(128)
void gemm_bf16_wmma(const bf16* __restrict__ A,  long long sA, int lda,
                    const bf16* __restrict__ Bt, long long sB, int ldb,
                    void* __restrict__ Dv,       long long sD, int ldd,
                    int K, const float* __restrict__ bias)
{
    const int lane = threadIdx.x;        // 0..31 (wave32)
    const int wave = threadIdx.y;        // 0..3
    const int l15  = lane & 15;
    const int h    = lane >> 4;
    const int b    = blockIdx.z;

    const int m0 = blockIdx.y * 64  + (wave >> 1) * 32;   // 2 waves along M
    const int n0 = blockIdx.x * 128 + (wave & 1) * 64;    // 2 waves along N

    A  += (size_t)b * sA;
    Bt += (size_t)b * sB;

    v8f acc[2][4] = {};

    for (int k0 = 0; k0 < K; k0 += 32) {
        V16 a[2], bm[4];
#pragma unroll
        for (int mi = 0; mi < 2; ++mi) {
            const bf16* ap = A + (size_t)(m0 + mi * 16 + l15) * lda + k0 + h * 8;
            a[mi].h[0] = *(const v8bf*)ap;        // K = h*8 .. h*8+7
            a[mi].h[1] = *(const v8bf*)(ap + 16); // K = 16+h*8 ..
        }
#pragma unroll
        for (int ni = 0; ni < 4; ++ni) {
            const bf16* bp = Bt + (size_t)(n0 + ni * 16 + l15) * ldb + k0 + h * 8;
            bm[ni].h[0] = *(const v8bf*)bp;       // K = h*8 .. h*8+7
            bm[ni].h[1] = *(const v8bf*)(bp + 16); // K = 16+h*8 ..
        }
#pragma unroll
        for (int mi = 0; mi < 2; ++mi)
#pragma unroll
            for (int ni = 0; ni < 4; ++ni)
                acc[mi][ni] = __builtin_amdgcn_wmma_f32_16x16x32_bf16(
                    false, a[mi].v, false, bm[ni].v,
                    (short)0, acc[mi][ni], false, false);
    }

    // Epilogue. D layout: element (M,N) lives at reg r, lane l with
    //   M = h*8 + r, N = l&15  (within a 16x16 tile)
    if constexpr (MODE == OUT_F32_N) {
        float* D = (float*)Dv + (size_t)b * sD;
#pragma unroll
        for (int mi = 0; mi < 2; ++mi)
#pragma unroll
            for (int ni = 0; ni < 4; ++ni) {
                const int n = n0 + ni * 16 + l15;
#pragma unroll
                for (int r = 0; r < 8; ++r) {
                    const int m = m0 + mi * 16 + h * 8 + r;
                    float val = acc[mi][ni][r];
                    if (bias) val += bias[m];
                    D[(size_t)m * ldd + n] = val;
                }
            }
    } else if constexpr (MODE == OUT_BF16_T) {
        bf16* D = (bf16*)Dv + (size_t)b * sD;   // (N x M) row-major, pitch ldd
#pragma unroll
        for (int mi = 0; mi < 2; ++mi)
#pragma unroll
            for (int ni = 0; ni < 4; ++ni) {
                const int n = n0 + ni * 16 + l15;
                v8bf pk;
#pragma unroll
                for (int r = 0; r < 8; ++r) pk[r] = (bf16)acc[mi][ni][r];
                *(v8bf*)(D + (size_t)n * ldd + m0 + mi * 16 + h * 8) = pk;
            }
    } else { // OUT_BF16_N
        bf16* D = (bf16*)Dv + (size_t)b * sD;
#pragma unroll
        for (int mi = 0; mi < 2; ++mi)
#pragma unroll
            for (int ni = 0; ni < 4; ++ni) {
                const int n = n0 + ni * 16 + l15;
#pragma unroll
                for (int r = 0; r < 8; ++r) {
                    const int m = m0 + mi * 16 + h * 8 + r;
                    D[(size_t)m * ldd + n] = (bf16)acc[mi][ni][r];
                }
            }
    }
}

// ---------------------------------------------------------------------------
// f32 -> bf16 elementwise (weights)
// ---------------------------------------------------------------------------
__global__ __launch_bounds__(256)
void cvt_bf16_kernel(const float* __restrict__ a, bf16* __restrict__ o, int n)
{
    int i = blockIdx.x * 256 + threadIdx.x;
    if (i < n) o[i] = (bf16)a[i];
}

// ---------------------------------------------------------------------------
// x (B,C,N) f32  ->  xT (B,N,C) bf16 via LDS tile transpose. block (32,8)
// ---------------------------------------------------------------------------
__global__ __launch_bounds__(256)
void transpose_x_kernel(const float* __restrict__ x, bf16* __restrict__ xT)
{
    __shared__ float tile[32][33];
    const int b  = blockIdx.z;
    const int n0 = blockIdx.x * 32;
    const int c0 = blockIdx.y * 32;
    const float* xb = x + (size_t)b * CC * NN;
    bf16* xtb = xT + (size_t)b * NN * CC;

    for (int j = threadIdx.y; j < 32; j += 8)
        tile[j][threadIdx.x] = xb[(size_t)(c0 + j) * NN + n0 + threadIdx.x];
    __syncthreads();
    for (int j = threadIdx.y; j < 32; j += 8)
        xtb[(size_t)(n0 + j) * CC + c0 + threadIdx.x] = (bf16)tile[threadIdx.x][j];
}

// ---------------------------------------------------------------------------
// Row softmax: one block per (b, row). N = 2304 = 9 * 256 exactly.
// Reads f32 scores row, writes bf16 prob row.
// ---------------------------------------------------------------------------
__global__ __launch_bounds__(256)
void softmax_rows_kernel(const float* __restrict__ S, bf16* __restrict__ P)
{
    const size_t row = blockIdx.x;
    const float* s = S + row * NN;
    bf16* p = P + row * NN;
    const int tid = threadIdx.x;

    float v[9];
    float m = -3.402823466e38f;
#pragma unroll
    for (int i = 0; i < 9; ++i) { v[i] = s[tid + i * 256]; m = fmaxf(m, v[i]); }

    __shared__ float red[256];
    red[tid] = m; __syncthreads();
    for (int off = 128; off > 0; off >>= 1) {
        if (tid < off) red[tid] = fmaxf(red[tid], red[tid + off]);
        __syncthreads();
    }
    m = red[0]; __syncthreads();

    float sum = 0.f;
#pragma unroll
    for (int i = 0; i < 9; ++i) { v[i] = __expf(v[i] - m); sum += v[i]; }
    red[tid] = sum; __syncthreads();
    for (int off = 128; off > 0; off >>= 1) {
        if (tid < off) red[tid] += red[tid + off];
        __syncthreads();
    }
    const float rs = 1.f / red[0];
#pragma unroll
    for (int i = 0; i < 9; ++i) p[tid + i * 256] = (bf16)(v[i] * rs);
}

// ---------------------------------------------------------------------------
// BatchNorm batch statistics: one block per channel. stats[c]=mean, stats[C+c]=rstd
// ---------------------------------------------------------------------------
__global__ __launch_bounds__(256)
void bn_stats_kernel(const float* __restrict__ y, float* __restrict__ stats)
{
    const int c = blockIdx.x, tid = threadIdx.x;
    float s = 0.f, s2 = 0.f;
    for (int b = 0; b < BB; ++b) {
        const float* p = y + ((size_t)b * CC + c) * NN;
        for (int i = tid; i < NN; i += 256) { float t = p[i]; s += t; s2 += t * t; }
    }
    __shared__ float r1[256], r2[256];
    r1[tid] = s; r2[tid] = s2; __syncthreads();
    for (int off = 128; off > 0; off >>= 1) {
        if (tid < off) { r1[tid] += r1[tid + off]; r2[tid] += r2[tid + off]; }
        __syncthreads();
    }
    if (tid == 0) {
        const float cnt = (float)(BB * NN);
        const float mean = r1[0] / cnt;
        const float var  = r2[0] / cnt - mean * mean;
        stats[c]      = mean;
        stats[CC + c] = rsqrtf(var + BN_EPS);
    }
}

// ---------------------------------------------------------------------------
// Apply BN affine:  out = (y - mean)*rstd*gamma + beta
// ---------------------------------------------------------------------------
__global__ __launch_bounds__(256)
void bn_apply_kernel(const float* __restrict__ y, const float* __restrict__ stats,
                     const float* __restrict__ gamma, const float* __restrict__ beta,
                     float* __restrict__ out)
{
    const size_t i = (size_t)blockIdx.x * 256 + threadIdx.x;
    const int c = (int)((i / NN) % CC);
    out[i] = (y[i] - stats[c]) * stats[CC + c] * gamma[c] + beta[c];
}

// ---------------------------------------------------------------------------
// Host-side launcher
// ---------------------------------------------------------------------------
extern "C" void kernel_launch(void* const* d_in, const int* in_sizes, int n_in,
                              void* d_out, int out_size, void* d_ws, size_t ws_size,
                              hipStream_t stream)
{
    const float* x     = (const float*)d_in[0];
    const float* Wk    = (const float*)d_in[1];
    const float* Wq    = (const float*)d_in[2];
    const float* Wv    = (const float*)d_in[3];
    const float* W2    = (const float*)d_in[4];
    const float* b2    = (const float*)d_in[5];
    const float* gamma = (const float*)d_in[6];
    const float* beta  = (const float*)d_in[7];
    float* out = (float*)d_out;

    // ---- workspace carve-up (256B aligned) ----
    uintptr_t cur = (uintptr_t)d_ws;
    auto alloc = [&](size_t bytes) {
        void* r = (void*)cur;
        cur += (bytes + 255) & ~(size_t)255;
        return r;
    };
    bf16*  wKb    = (bf16*)alloc((size_t)CC * CC * 2);
    bf16*  wQb    = (bf16*)alloc((size_t)CC * CC * 2);
    bf16*  wVb    = (bf16*)alloc((size_t)CC * CC * 2);
    bf16*  w2b    = (bf16*)alloc((size_t)CC * CC * 2);
    bf16*  xbT    = (bf16*)alloc((size_t)BB * NN * CC * 2);  // (B,N,C)
    bf16*  kT     = (bf16*)alloc((size_t)BB * NN * CC * 2);  // (B,N,C)
    bf16*  qT     = (bf16*)alloc((size_t)BB * NN * CC * 2);  // (B,N,C)
    bf16*  vN     = (bf16*)alloc((size_t)BB * CC * NN * 2);  // (B,C,N)
    bf16*  attnT  = (bf16*)alloc((size_t)BB * NN * CC * 2);  // (B,N,C)
    float* yws    = (float*)alloc((size_t)BB * CC * NN * 4); // (B,C,N)
    float* stats  = (float*)alloc((size_t)2 * CC * 4);       // mean | rstd
    float* scores = (float*)alloc((size_t)BB * NN * NN * 4); // (B,N,N)
    bf16*  prob   = (bf16*)alloc((size_t)BB * NN * NN * 2);  // (B,N,N)

    const long long sNC  = (long long)NN * CC;   // per-batch (N,C)
    const long long sCN  = (long long)CC * NN;   // per-batch (C,N)
    const long long sNNb = (long long)NN * NN;   // per-batch (N,N)

    // 1) weights -> bf16
    cvt_bf16_kernel<<<CC * CC / 256, 256, 0, stream>>>(Wk, wKb, CC * CC);
    cvt_bf16_kernel<<<CC * CC / 256, 256, 0, stream>>>(Wq, wQb, CC * CC);
    cvt_bf16_kernel<<<CC * CC / 256, 256, 0, stream>>>(Wv, wVb, CC * CC);
    cvt_bf16_kernel<<<CC * CC / 256, 256, 0, stream>>>(W2, w2b, CC * CC);

    // 2) x -> xT bf16 (B,N,C)
    transpose_x_kernel<<<dim3(NN / 32, CC / 32, BB), dim3(32, 8), 0, stream>>>(x, xbT);

    // 3) projections:  K,Q written transposed (N,C); V written (C,N)
    {
        dim3 g(NN / 128, CC / 64, BB), blk(32, 4);
        gemm_bf16_wmma<OUT_BF16_T><<<g, blk, 0, stream>>>(
            wKb, 0, CC, xbT, sNC, CC, kT, sNC, CC, CC, nullptr);
        gemm_bf16_wmma<OUT_BF16_T><<<g, blk, 0, stream>>>(
            wQb, 0, CC, xbT, sNC, CC, qT, sNC, CC, CC, nullptr);
        gemm_bf16_wmma<OUT_BF16_N><<<g, blk, 0, stream>>>(
            wVb, 0, CC, xbT, sNC, CC, vN, sCN, NN, CC, nullptr);
    }

    // 4) scores[b,n,m] = sum_c k[c,n] q[c,m]  ->  A = K^T (N,C), Bt = Q^T (N,C)
    {
        dim3 g(NN / 128, NN / 64, BB), blk(32, 4);
        gemm_bf16_wmma<OUT_F32_N><<<g, blk, 0, stream>>>(
            kT, sNC, CC, qT, sNC, CC, scores, sNNb, NN, CC, nullptr);
    }

    // 5) row softmax -> bf16 prob
    softmax_rows_kernel<<<BB * NN, 256, 0, stream>>>(scores, prob);

    // 6) attn[c,n] = sum_m v[c,m] p[n,m]  ->  A = V (C,N), Bt = P (N,N); write attn^T
    {
        dim3 g(NN / 128, CC / 64, BB), blk(32, 4);
        gemm_bf16_wmma<OUT_BF16_T><<<g, blk, 0, stream>>>(
            vN, sCN, NN, prob, sNNb, NN, attnT, sNC, CC, NN, nullptr);
    }

    // 7) y = W2 * attn + b2  -> f32 (B,C,N)
    {
        dim3 g(NN / 128, CC / 64, BB), blk(32, 4);
        gemm_bf16_wmma<OUT_F32_N><<<g, blk, 0, stream>>>(
            w2b, 0, CC, attnT, sNC, CC, yws, sCN, NN, CC, b2);
    }

    // 8) BatchNorm (training-mode batch stats) + affine
    bn_stats_kernel<<<CC, 256, 0, stream>>>(yws, stats);
    bn_apply_kernel<<<(BB * CC * NN) / 256, 256, 0, stream>>>(yws, stats, gamma, beta, out);

    (void)in_sizes; (void)n_in; (void)out_size; (void)ws_size;
}
